// CustomLSTMModel_72086731096306
// MI455X (gfx1250) — matrix operand order repaired and checked by reading, not compiled, not run on hardware
//
#include <hip/hip_runtime.h>

#define B_ 64
#define S_ 512
#define D_ 512
#define H_ 1024
#define KCX 16   // D/32  k-chunks over input part
#define KCH 32   // H/32  k-chunks over recurrent part

typedef __attribute__((ext_vector_type(16))) __bf16 v16bf;
typedef __attribute__((ext_vector_type(8)))  __bf16 v8bf;
typedef __attribute__((ext_vector_type(8)))  float  v8f;

// Scheduling-control hints (standard AMDGPU intrinsics; no-op fallback).
#if __has_builtin(__builtin_amdgcn_sched_group_barrier)
#define SGB(mask, n, sync) __builtin_amdgcn_sched_group_barrier((mask), (n), (sync))
#else
#define SGB(mask, n, sync)
#endif

// fp32 -> bf16 round-to-nearest-even
__device__ __forceinline__ unsigned short f2bf(float f) {
    unsigned int u = __float_as_uint(f);
    u += 0x7fffu + ((u >> 16) & 1u);
    return (unsigned short)(u >> 16);
}

__device__ __forceinline__ float sigmoid_f(float x) {
    x = fminf(fmaxf(x, -30.0f), 30.0f);
    return 1.0f / (1.0f + __expf(-x));
}
__device__ __forceinline__ float tanh_f(float x) {
    x = fminf(fmaxf(x, -15.0f), 15.0f);
    float e = __expf(-2.0f * x);
    return (1.0f - e) / (1.0f + e);
}

// A-fragment (16x32 bf16, wave32 layout): lane: m = L&15, khalf = L>>4 holds
// K = khalf*8 + {0..7} and 16 + khalf*8 + {0..7}: two contiguous 16B loads.
__device__ __forceinline__ v16bf ldA(const unsigned short* __restrict__ row, int kbase) {
    v8bf lo = *(const v8bf*)(row + kbase);
    v8bf hi = *(const v8bf*)(row + kbase + 16);
    return __builtin_shufflevector(lo, hi, 0,1,2,3,4,5,6,7,8,9,10,11,12,13,14,15);
}
__device__ __forceinline__ v16bf ldB(const unsigned short* __restrict__ p) {
    return *(const v16bf*)p;
}

// Distance-2 software-pipelined K-loop: chunks kc and kc+1 are resident in
// registers; iteration kc issues the 10 VMEM reads for chunk kc+2, then the 4
// WMMAs for chunk kc. The adjacent VMEM group is therefore two chunks ahead
// of its consumer, so no schedule can drain loadcnt to 0 before a WMMA —
// a full chunk of load latency always hides under the matrix ops.
template <int NC>
__device__ __forceinline__ void gemm_part(const unsigned short* __restrict__ row, int khalf,
                                          const unsigned short* __restrict__ bp0,
                                          const unsigned short* __restrict__ bp1,
                                          const unsigned short* __restrict__ bp2,
                                          const unsigned short* __restrict__ bp3,
                                          v8f& acc0, v8f& acc1, v8f& acc2, v8f& acc3)
{
    v16bf a0  = ldA(row, khalf * 8);
    v16bf b00 = ldB(bp0), b10 = ldB(bp1), b20 = ldB(bp2), b30 = ldB(bp3);
    v16bf a1  = ldA(row, 32 + khalf * 8);
    v16bf b01 = ldB(bp0 + 512), b11 = ldB(bp1 + 512), b21 = ldB(bp2 + 512), b31 = ldB(bp3 + 512);
#pragma unroll
    for (int kc = 0; kc < NC; ++kc) {
        v16bf an, b0n, b1n, b2n, b3n;
        if (kc + 2 < NC) {
            an  = ldA(row, (kc + 2) * 32 + khalf * 8);
            b0n = ldB(bp0 + (kc + 2) * 512);
            b1n = ldB(bp1 + (kc + 2) * 512);
            b2n = ldB(bp2 + (kc + 2) * 512);
            b3n = ldB(bp3 + (kc + 2) * 512);
        }
        acc0 = __builtin_amdgcn_wmma_f32_16x16x32_bf16(false, a0, false, b00, (short)0, acc0, false, false);
        acc1 = __builtin_amdgcn_wmma_f32_16x16x32_bf16(false, a0, false, b10, (short)0, acc1, false, false);
        acc2 = __builtin_amdgcn_wmma_f32_16x16x32_bf16(false, a0, false, b20, (short)0, acc2, false, false);
        acc3 = __builtin_amdgcn_wmma_f32_16x16x32_bf16(false, a0, false, b30, (short)0, acc3, false, false);
        // per-chunk pattern: 10 VMEM reads (chunk kc+2) then 4 matrix ops (chunk kc)
        SGB(0x0020, 10, 0);   // VMEM read group
        SGB(0x0008, 4, 0);    // MFMA/WMMA group
        a0 = a1;  b00 = b01; b10 = b11; b20 = b21; b30 = b31;
        a1 = an;  b01 = b0n; b11 = b1n; b21 = b2n; b31 = b3n;
    }
}

// ---------------- preprocessing kernels ----------------

__global__ void k_cvt_inputs(const float* __restrict__ in, unsigned short* __restrict__ xb) {
    const int N = B_ * S_ * D_;
    for (int i = blockIdx.x * blockDim.x + threadIdx.x; i < N; i += gridDim.x * blockDim.x)
        xb[i] = f2bf(in[i]);
}

// Pack a [D or H, H] fp32 gate weight set into wave32 B-fragment order:
// dst[(g*64+ct)*KC*512 + kc*512 + lane*16 + e], lane: n=lane&15, khalf=lane>>4,
// src k = kc*32 + khalf*8 + (e&7) + (e>=8 ? 16 : 0), col = ct*16 + n.
__global__ void k_pack_w(const float* __restrict__ W0, const float* __restrict__ W1,
                         const float* __restrict__ W2, const float* __restrict__ W3,
                         unsigned short* __restrict__ dst, int KC) {
    const int total = 256 * KC * 512;
    for (int idx = blockIdx.x * blockDim.x + threadIdx.x; idx < total; idx += gridDim.x * blockDim.x) {
        int e   = idx & 511;
        int kc  = (idx >> 9) % KC;
        int gct = idx / (512 * KC);
        int g = gct >> 6, ct = gct & 63;
        int lane = e >> 4;
        int n = lane & 15, khalf = lane >> 4;
        int k   = kc * 32 + khalf * 8 + (e & 7) + ((e & 8) << 1);
        int col = ct * 16 + n;
        const float* W = (g == 0) ? W0 : (g == 1) ? W1 : (g == 2) ? W2 : W3;
        dst[idx] = f2bf(W[(size_t)k * H_ + col]);
    }
}

__global__ void k_init_state(const float* __restrict__ H0, const float* __restrict__ C0,
                             unsigned short* __restrict__ Hbuf0, float* __restrict__ Cbuf) {
    int i = blockIdx.x * blockDim.x + threadIdx.x;
    if (i < B_ * H_) { Hbuf0[i] = f2bf(H0[i]); Cbuf[i] = C0[i]; }
}

__global__ void k_init_pred(float* __restrict__ pred, const float* __restrict__ fcb) {
    int i = blockIdx.x * blockDim.x + threadIdx.x;
    if (i < B_ * S_) pred[i] = fcb[0];
}

// ---------------- fused LSTM step ----------------
// Block = one h column-tile (ct = blockIdx.x, 64 blocks). The 4 waves of a
// block cover batch tiles rt=0..3 and read IDENTICAL B fragments -> they hit
// in the CU L0 / WGP cache, cutting L2 weight traffic ~4x. Each wave holds 4
// gate accumulators (i,f,o,c); gate columns land in the same lane/VGPR slot,
// so the LSTM cell math is pure per-lane VALU.

__global__ void __launch_bounds__(128)
k_lstm_step(const unsigned short* __restrict__ xb,
            const unsigned short* __restrict__ WxF,
            const unsigned short* __restrict__ WhF,
            const unsigned short* __restrict__ Hcur,
            unsigned short* __restrict__ Hnxt,
            float* __restrict__ Cbuf,
            const float* __restrict__ b_i, const float* __restrict__ b_f,
            const float* __restrict__ b_o, const float* __restrict__ b_c,
            const float* __restrict__ fcW,
            float* __restrict__ pred, float* __restrict__ Hf, float* __restrict__ Cf,
            int t, int isLast)
{
    const int lane  = threadIdx.x & 31;
    const int rt    = threadIdx.x >> 5;   // batch tile: one per wave
    const int ct    = blockIdx.x;         // h tile: one per block (shared B)
    const int m     = lane & 15;
    const int khalf = lane >> 4;

    const unsigned short* xrow = xb   + ((size_t)(rt * 16 + m) * S_ + t) * D_;
    const unsigned short* hrow = Hcur + (size_t)(rt * 16 + m) * H_;

    v8f acc0 = {}, acc1 = {}, acc2 = {}, acc3 = {};

    // ---- x_t @ Wx (K = 512) ----
    gemm_part<KCX>(xrow, khalf,
                   WxF + (size_t)(0 * 64 + ct) * KCX * 512 + lane * 16,
                   WxF + (size_t)(1 * 64 + ct) * KCX * 512 + lane * 16,
                   WxF + (size_t)(2 * 64 + ct) * KCX * 512 + lane * 16,
                   WxF + (size_t)(3 * 64 + ct) * KCX * 512 + lane * 16,
                   acc0, acc1, acc2, acc3);
    // ---- H_t @ Wh (K = 1024) ----
    gemm_part<KCH>(hrow, khalf,
                   WhF + (size_t)(0 * 64 + ct) * KCH * 512 + lane * 16,
                   WhF + (size_t)(1 * 64 + ct) * KCH * 512 + lane * 16,
                   WhF + (size_t)(2 * 64 + ct) * KCH * 512 + lane * 16,
                   WhF + (size_t)(3 * 64 + ct) * KCH * 512 + lane * 16,
                   acc0, acc1, acc2, acc3);

    // ---- gates + state update (C/D layout: element v -> M = v + 8*khalf, N = lane&15) ----
    const int n = lane & 15;
    const int h = ct * 16 + n;
    const float bib = b_i[h], bfb = b_f[h], bob = b_o[h], bcb = b_c[h];
    const float fw  = fcW[h];

    float partial[8];
#pragma unroll
    for (int v = 0; v < 8; ++v) {
        const int b = rt * 16 + v + 8 * khalf;
        const size_t idx = (size_t)b * H_ + h;
        float gi = sigmoid_f(acc0[v] + bib);
        float gf = sigmoid_f(acc1[v] + bfb);
        float go = sigmoid_f(acc2[v] + bob);
        float gc = tanh_f   (acc3[v] + bcb);
        float cn = gf * Cbuf[idx] + gi * gc;
        float hn = go * tanh_f(cn);
        Cbuf[idx] = cn;
        Hnxt[idx] = f2bf(hn);
        if (isLast) { Hf[idx] = hn; Cf[idx] = cn; }
        partial[v] = hn * fw;
    }

    // butterfly-reduce over the 16 N-lanes of this half (xor masks never flip bit 4)
#pragma unroll
    for (int off = 1; off < 16; off <<= 1) {
#pragma unroll
        for (int v = 0; v < 8; ++v)
            partial[v] += __shfl_xor(partial[v], off, 32);
    }
    if (n == 0) {
#pragma unroll
        for (int v = 0; v < 8; ++v) {
            const int b = rt * 16 + v + 8 * khalf;
            atomicAdd(&pred[(size_t)b * S_ + t], partial[v]);
        }
    }
}

// ---------------- host launch ----------------

extern "C" void kernel_launch(void* const* d_in, const int* in_sizes, int n_in,
                              void* d_out, int out_size, void* d_ws, size_t ws_size,
                              hipStream_t stream)
{
    (void)in_sizes; (void)n_in; (void)out_size; (void)ws_size;

    const float* inputs = (const float*)d_in[0];
    const float* H0     = (const float*)d_in[1];
    const float* C0     = (const float*)d_in[2];
    const float* Wx[4]  = {(const float*)d_in[3], (const float*)d_in[6], (const float*)d_in[9],  (const float*)d_in[12]};
    const float* Wh[4]  = {(const float*)d_in[4], (const float*)d_in[7], (const float*)d_in[10], (const float*)d_in[13]};
    const float* bia[4] = {(const float*)d_in[5], (const float*)d_in[8], (const float*)d_in[11], (const float*)d_in[14]};
    const float* fcW    = (const float*)d_in[15];
    const float* fcb    = (const float*)d_in[16];

    // workspace carve-up (all offsets 256B-aligned)
    char* ws = (char*)d_ws;
    unsigned short* xb   = (unsigned short*)(ws);               // 32 MB: inputs in bf16
    unsigned short* WxF  = (unsigned short*)(ws + 33554432);    //  4 MB: packed Wx fragments
    unsigned short* WhF  = (unsigned short*)(ws + 37748736);    //  8 MB: packed Wh fragments
    unsigned short* Hbuf = (unsigned short*)(ws + 46137344);    // 256 KB: H ping-pong (bf16)
    float*          Cbuf = (float*)         (ws + 46399488);    // 256 KB: C state (f32)

    float* pred = (float*)d_out;          // [B, S, 1]
    float* Hf   = pred + B_ * S_;         // [B, H]
    float* Cf   = Hf   + B_ * H_;         // [B, H]

    k_cvt_inputs<<<2048, 256, 0, stream>>>(inputs, xb);
    k_pack_w<<<2048, 256, 0, stream>>>(Wx[0], Wx[1], Wx[2], Wx[3], WxF, KCX);
    k_pack_w<<<2048, 256, 0, stream>>>(Wh[0], Wh[1], Wh[2], Wh[3], WhF, KCH);
    k_init_state<<<(B_ * H_ + 255) / 256, 256, 0, stream>>>(H0, C0, Hbuf, Cbuf);
    k_init_pred<<<(B_ * S_ + 255) / 256, 256, 0, stream>>>(pred, fcb);

    for (int t = 0; t < S_; ++t) {
        const unsigned short* hc = Hbuf + (size_t)(t & 1) * (B_ * H_);
        unsigned short*       hn = Hbuf + (size_t)((t + 1) & 1) * (B_ * H_);
        k_lstm_step<<<64, 128, 0, stream>>>(xb, WxF, WhF, hc, hn, Cbuf,
                                            bia[0], bia[1], bia[2], bia[3], fcW,
                                            pred, Hf, Cf, t, (t == S_ - 1) ? 1 : 0);
    }
}
